// SpatialTransformer_6880537608479
// MI455X (gfx1250) — matrix operand order repaired and checked by reading, not compiled, not run on hardware
//
#include <hip/hip_runtime.h>
#include <hip/hip_bf16.h>

typedef __attribute__((ext_vector_type(16))) _Float16 v16h;
typedef __attribute__((ext_vector_type(8)))  _Float16 v8h;
typedef __attribute__((ext_vector_type(4)))  _Float16 v4h;
typedef __attribute__((ext_vector_type(8)))  float    v8f;
typedef __attribute__((ext_vector_type(4)))  float    v4f;

#define N_NODES 16384
#define DMODEL  512
#define BSZ     16
#define MCOND   256
#define NHEAD   8
#define DHEAD   64
#define NLAYER  4
#define NEDGE   262144
#define LN_EPS  1e-5f

// ---------------- WMMA GEMM (f32 in/out, f16 compute, f32 accumulate) ----------------
// Requirements guaranteed by all call sites below:
//   * M % 128 == 0, K % 32 == 0 (no M/K guards needed)
//   * all leading dims and batch strides are multiples of 4 floats (16B vector loads)
//   * B-tile reads may overhang N logically (N=64 case) but stay inside allocated
//     workspace; epilogue guards stores with n < N.
#define BM 128
#define BN 128
#define BK 32
#define LDK 40   // 40 halves = 80 bytes per row -> 16B aligned fragment loads

// A fragment, 16x32 f16 (ISA 7.12.2): lane L holds row M=L%16;
// elements 0..7 -> K = base+0..7, elements 8..15 -> K = base+16..23, base = 8*(L>=16)
__device__ __forceinline__ v16h frag_load_a(const _Float16* lds, int row) {
  const int lane = threadIdx.x & 31;
  const _Float16* p = lds + (row + (lane & 15)) * LDK + ((lane >> 4) << 3);
  v8h lo = *(const v8h*)(p);
  v8h hi = *(const v8h*)(p + 16);
  v16h r;
#pragma unroll
  for (int i = 0; i < 8; ++i) { r[i] = lo[i]; r[i + 8] = hi[i]; }
  return r;
}

// B fragment, 32x16 f16: lane L holds col N=L%16; elements 0..7 -> K = base+0..7, elements 8..15 -> K = base+16..23, base = 8*(L>=16)
// (LDS tile stored [N][K])
__device__ __forceinline__ v16h frag_load_b(const _Float16* lds, int row) {
  const int lane = threadIdx.x & 31;
  const _Float16* p = lds + (row + (lane & 15)) * LDK + ((lane >> 4) << 3);
  v8h lo = *(const v8h*)(p);
  v8h hi = *(const v8h*)(p + 16);
  v16h r;
#pragma unroll
  for (int i = 0; i < 8; ++i) { r[i] = lo[i]; r[i + 8] = hi[i]; }
  return r;
}

// C = A @ B (+bias) (+residual).  A: MxK row-major (lda), B: KxN (ldb) or NxK if TRANSB,
// C: MxN (ldc). Batch dim z = z1*Z2+z2 with independent strides for A/B/C (residual uses C's).
template<bool TRANSB>
__global__ __launch_bounds__(256)
void wmma_gemm_kernel(const float* __restrict__ A, const float* __restrict__ B,
                      float* __restrict__ C,
                      const float* __restrict__ bias,
                      const float* __restrict__ residual,
                      int M, int N, int K, int lda, int ldb, int ldc, int Z2,
                      long long sA1, long long sA2,
                      long long sB1, long long sB2,
                      long long sC1, long long sC2)
{
  __shared__ __align__(16) _Float16 As[2 * BM * LDK];
  __shared__ __align__(16) _Float16 Bs[2 * BN * LDK];

  const int tid = threadIdx.x;
  const int z  = blockIdx.z;
  const int z1 = z / Z2, z2 = z - z1 * Z2;
  A += (long long)z1 * sA1 + (long long)z2 * sA2;
  B += (long long)z1 * sB1 + (long long)z2 * sB2;
  C += (long long)z1 * sC1 + (long long)z2 * sC2;
  const float* R = residual ? residual + (long long)z1 * sC1 + (long long)z2 * sC2 : nullptr;

  const int mBase = blockIdx.y * BM;
  const int nBase = blockIdx.x * BN;

  const int wid = tid >> 5;
  const int wmOff = (wid >> 2) * 64;  // 2 wave-rows of 64
  const int wnOff = (wid & 3) * 32;   // 4 wave-cols of 32

  // staging geometry (vectorized float4)
  const int arBase = tid >> 3;         // 0..31   (row group; rows arBase+32*it)
  const int ac     = (tid & 7) << 2;   // 0..28   (k offset within tile)
  const int bkBase = tid >> 5;         // 0..7    (k rows for non-trans B)
  const int bn4    = (tid & 31) << 2;  // 0..124  (n offset for non-trans B)

  v4f ra[4], rb[4];

  auto g_load = [&](int k0) {
#pragma unroll
    for (int it = 0; it < 4; ++it)
      ra[it] = *(const v4f*)(A + (long long)(mBase + arBase + 32 * it) * lda + k0 + ac);
    if (TRANSB) {
#pragma unroll
      for (int it = 0; it < 4; ++it)
        rb[it] = *(const v4f*)(B + (long long)(nBase + arBase + 32 * it) * ldb + k0 + ac);
    } else {
#pragma unroll
      for (int it = 0; it < 4; ++it)
        rb[it] = *(const v4f*)(B + (long long)(k0 + bkBase + 8 * it) * ldb + nBase + bn4);
    }
    // prefetch one K-tile further ahead (CDNA5 global_prefetch_b8)
    if (k0 + BK < K) {
      __builtin_prefetch(A + (long long)(mBase + arBase) * lda + k0 + BK + ac, 0, 1);
      if (TRANSB)
        __builtin_prefetch(B + (long long)(nBase + arBase) * ldb + k0 + BK + ac, 0, 1);
      else
        __builtin_prefetch(B + (long long)(k0 + BK + bkBase) * ldb + nBase + bn4, 0, 1);
    }
  };

  auto s_store = [&](int buf) {
    _Float16* Ab = As + buf * (BM * LDK);
    _Float16* Bb = Bs + buf * (BN * LDK);
#pragma unroll
    for (int it = 0; it < 4; ++it) {
      v4h hv;
#pragma unroll
      for (int j = 0; j < 4; ++j) hv[j] = (_Float16)ra[it][j];
      *(v4h*)(Ab + (arBase + 32 * it) * LDK + ac) = hv;
    }
    if (TRANSB) {
#pragma unroll
      for (int it = 0; it < 4; ++it) {
        v4h hv;
#pragma unroll
        for (int j = 0; j < 4; ++j) hv[j] = (_Float16)rb[it][j];
        *(v4h*)(Bb + (arBase + 32 * it) * LDK + ac) = hv;
      }
    } else {
#pragma unroll
      for (int it = 0; it < 4; ++it)
#pragma unroll
        for (int j = 0; j < 4; ++j)
          Bb[(bn4 + j) * LDK + bkBase + 8 * it] = (_Float16)rb[it][j];
    }
  };

  const v8f vzero = {0.f, 0.f, 0.f, 0.f, 0.f, 0.f, 0.f, 0.f};
  v8f acc[4][2];
#pragma unroll
  for (int i = 0; i < 4; ++i)
#pragma unroll
    for (int j = 0; j < 2; ++j) acc[i][j] = vzero;

  // software pipeline: global loads for tile k+1 issued before WMMAs on tile k,
  // LDS stores after WMMAs, one barrier per K-step, double-buffered LDS.
  g_load(0);
  s_store(0);
  __syncthreads();

  const int nk = K >> 5;
  for (int kt = 0; kt < nk; ++kt) {
    const int cur = kt & 1;
    if (kt + 1 < nk) g_load((kt + 1) * BK);

    const _Float16* Ab = As + cur * (BM * LDK);
    const _Float16* Bb = Bs + cur * (BN * LDK);
    v16h aF[4], bF[2];
#pragma unroll
    for (int mi = 0; mi < 4; ++mi) aF[mi] = frag_load_a(Ab, wmOff + mi * 16);
#pragma unroll
    for (int ni = 0; ni < 2; ++ni) bF[ni] = frag_load_b(Bb, wnOff + ni * 16);
#pragma unroll
    for (int mi = 0; mi < 4; ++mi)
#pragma unroll
      for (int ni = 0; ni < 2; ++ni)
        acc[mi][ni] = __builtin_amdgcn_wmma_f32_16x16x32_f16(
            false, aF[mi], false, bF[ni], (short)0, acc[mi][ni], false, false);

    if (kt + 1 < nk) s_store((kt + 1) & 1);
    __syncthreads();
  }

  // epilogue: D layout -> M = r + 8*(lane>=16), N = lane&15
  const int lane = tid & 31;
  const int mSub = (lane >> 4) * 8;
  const int nSub = lane & 15;
#pragma unroll
  for (int mi = 0; mi < 4; ++mi) {
#pragma unroll
    for (int ni = 0; ni < 2; ++ni) {
      int n = nBase + wnOff + ni * 16 + nSub;
      if (n >= N) continue;
      float bv = bias ? bias[n] : 0.f;
#pragma unroll
      for (int r8 = 0; r8 < 8; ++r8) {
        int m = mBase + wmOff + mi * 16 + mSub + r8;
        float v = acc[mi][ni][r8] + bv;
        if (R) v += R[(long long)m * ldc + n];
        C[(long long)m * ldc + n] = v;
      }
    }
  }
}

// ---------------- BatchNorm (column stats over N rows) ----------------
__global__ void bn_stats_kernel(const float* __restrict__ x, float* sum, float* sumsq) {
  int c0 = threadIdx.x, c1 = threadIdx.x + 256;
  long long rowBase = (long long)blockIdx.x * 64;
  float s0 = 0, s1 = 0, q0 = 0, q1 = 0;
  for (int r = 0; r < 64; ++r) {
    const float* p = x + (rowBase + r) * DMODEL;
    float v0 = p[c0], v1 = p[c1];
    s0 += v0; q0 += v0 * v0; s1 += v1; q1 += v1 * v1;
  }
  atomicAdd(&sum[c0], s0); atomicAdd(&sumsq[c0], q0);
  atomicAdd(&sum[c1], s1); atomicAdd(&sumsq[c1], q1);
}

__global__ void bn_apply_kernel(const float* __restrict__ x, const float* sum,
                                const float* sumsq, const float* w, const float* b,
                                float* __restrict__ out) {
  long long idx = (long long)blockIdx.x * 256 + threadIdx.x;
  int c = idx & (DMODEL - 1);
  float mu = sum[c] * (1.f / N_NODES);
  float var = sumsq[c] * (1.f / N_NODES) - mu * mu;
  out[idx] = (x[idx] - mu) * rsqrtf(var + LN_EPS) * w[c] + b[c];
}

// ---------------- GCN degree / aggregation ----------------
__global__ void degree_kernel(const long long* __restrict__ ei, float* deg) {
  int e = blockIdx.x * 256 + threadIdx.x;
  if (e < NEDGE) atomicAdd(&deg[(int)ei[NEDGE + e]], 1.f);
}

__global__ void dinv_kernel(const float* deg, float* dinv) {
  int i = blockIdx.x * 256 + threadIdx.x;
  if (i < N_NODES) dinv[i] = rsqrtf(deg[i] + 1.0f);   // +1 self-loop
}

__global__ void gcn_scatter_kernel(const float* __restrict__ hsrc,
                                   const long long* __restrict__ ei,
                                   const float* __restrict__ dinv,
                                   float* __restrict__ out) {
  int e = blockIdx.x;
  int s, d;
  if (e < NEDGE) { s = (int)ei[e]; d = (int)ei[NEDGE + e]; }
  else           { s = d = e - NEDGE; }
  float norm = dinv[s] * dinv[d];
  const float* src = hsrc + (long long)s * DMODEL;
  float* dst = out + (long long)d * DMODEL;
  for (int c = threadIdx.x; c < DMODEL; c += 256)
    atomicAdd(&dst[c], src[c] * norm);
}

__global__ void add_bias_kernel(float* __restrict__ out, const float* __restrict__ bias) {
  long long idx = (long long)blockIdx.x * 256 + threadIdx.x;
  out[idx] += bias[idx & (DMODEL - 1)];
}

// ---------------- LayerNorm: one wave32 per row of 512 ----------------
__global__ void ln_kernel(const float* __restrict__ x, const float* __restrict__ w,
                          const float* __restrict__ b, float* __restrict__ out) {
  int lane = threadIdx.x & 31;
  int row = blockIdx.x * 8 + (threadIdx.x >> 5);
  const float* p = x + (long long)row * DMODEL;
  float v[16]; float s = 0.f, q = 0.f;
#pragma unroll
  for (int i = 0; i < 16; ++i) { v[i] = p[lane + i * 32]; s += v[i]; q += v[i] * v[i]; }
#pragma unroll
  for (int off = 16; off; off >>= 1) { s += __shfl_xor(s, off, 32); q += __shfl_xor(q, off, 32); }
  float mu = s * (1.f / DMODEL);
  float var = q * (1.f / DMODEL) - mu * mu;
  float rs = rsqrtf(var + LN_EPS);
  float* o = out + (long long)row * DMODEL;
#pragma unroll
  for (int i = 0; i < 16; ++i) { int c = lane + i * 32; o[c] = (v[i] - mu) * rs * w[c] + b[c]; }
}

// ---------------- Softmax over 256 keys (scale 1/sqrt(64)=0.125), one wave per row ----------------
__global__ void softmax_kernel(float* __restrict__ att) {
  int lane = threadIdx.x & 31;
  long long row = (long long)blockIdx.x * 8 + (threadIdx.x >> 5);
  float* p = att + row * MCOND;
  float v[8]; float mx = -1e30f;
#pragma unroll
  for (int i = 0; i < 8; ++i) { v[i] = p[lane + i * 32] * 0.125f; mx = fmaxf(mx, v[i]); }
#pragma unroll
  for (int off = 16; off; off >>= 1) mx = fmaxf(mx, __shfl_xor(mx, off, 32));
  float s = 0.f;
#pragma unroll
  for (int i = 0; i < 8; ++i) { v[i] = expf(v[i] - mx); s += v[i]; }
#pragma unroll
  for (int off = 16; off; off >>= 1) s += __shfl_xor(s, off, 32);
  float inv = 1.f / s;
#pragma unroll
  for (int i = 0; i < 8; ++i) p[lane + i * 32] = v[i] * inv;
}

// ---------------- GEGLU: f = val * gelu_exact(gate) ----------------
__global__ void geglu_kernel(const float* __restrict__ u, float* __restrict__ f) {
  long long idx = (long long)blockIdx.x * 256 + threadIdx.x;   // N * 2048
  long long row = idx >> 11;
  int col = idx & 2047;
  const float* up = u + row * 4096;
  float val = up[col];
  float g = up[2048 + col];
  float ge = 0.5f * g * (1.f + erff(g * 0.70710678118654752f));
  f[idx] = val * ge;
}

// ---------------- Host-side helper ----------------
static void gemm(hipStream_t st, bool transb,
                 const float* A, const float* B, float* C,
                 const float* bias, const float* residual,
                 int M, int N, int K, int lda, int ldb, int ldc,
                 int Z1 = 1, int Z2 = 1,
                 long long sA1 = 0, long long sA2 = 0,
                 long long sB1 = 0, long long sB2 = 0,
                 long long sC1 = 0, long long sC2 = 0) {
  dim3 grid((N + BN - 1) / BN, (M + BM - 1) / BM, Z1 * Z2);
  if (transb)
    wmma_gemm_kernel<true><<<grid, 256, 0, st>>>(A, B, C, bias, residual, M, N, K,
                                                 lda, ldb, ldc, Z2, sA1, sA2, sB1, sB2, sC1, sC2);
  else
    wmma_gemm_kernel<false><<<grid, 256, 0, st>>>(A, B, C, bias, residual, M, N, K,
                                                  lda, ldb, ldc, Z2, sA1, sA2, sB1, sB2, sC1, sC2);
}

extern "C" void kernel_launch(void* const* d_in, const int* in_sizes, int n_in,
                              void* d_out, int out_size, void* d_ws, size_t ws_size,
                              hipStream_t stream) {
  const float*     x         = (const float*)d_in[0];
  const long long* ei        = (const long long*)d_in[1];
  const float*     cond      = (const float*)d_in[2];
  const float*     bn_w      = (const float*)d_in[3];
  const float*     bn_b      = (const float*)d_in[4];
  const float*     gcn_in_w  = (const float*)d_in[5];
  const float*     gcn_in_b  = (const float*)d_in[6];
  const float*     gcn_out_w = (const float*)d_in[7];
  const float*     gcn_out_b = (const float*)d_in[8];
  const float*     Wq        = (const float*)d_in[9];
  const float*     Wk        = (const float*)d_in[10];
  const float*     Wv        = (const float*)d_in[11];
  const float*     Wo        = (const float*)d_in[12];
  const float*     ln2_w     = (const float*)d_in[13];
  const float*     ln2_b     = (const float*)d_in[14];
  const float*     ln3_w     = (const float*)d_in[15];
  const float*     ln3_b     = (const float*)d_in[16];
  const float*     geglu_w   = (const float*)d_in[17];
  const float*     geglu_b   = (const float*)d_in[18];
  const float*     ffo_w     = (const float*)d_in[19];
  const float*     ffo_b     = (const float*)d_in[20];
  float* out = (float*)d_out;
  (void)in_sizes; (void)n_in; (void)out_size; (void)ws_size;

  // workspace layout (floats)
  float* W = (float*)d_ws;
  size_t off = 0;
  float* colsum = W + off; off += 512;
  float* colsq  = W + off; off += 512;
  float* deg    = W + off; off += N_NODES;
  float* dinv   = W + off; off += N_NODES;
  const size_t ND = (size_t)N_NODES * DMODEL;
  float* h   = W + off; off += ND;                         // running state
  float* t0  = W + off; off += ND;                         // LN output
  float* t1  = W + off; off += ND;                         // GEMM temp / attn out
  float* q   = W + off; off += ND;                         // queries
  float* kb  = W + off; off += (size_t)BSZ * MCOND * DMODEL;
  float* vb  = W + off; off += (size_t)BSZ * MCOND * DMODEL;
  float* att = W + off; off += (size_t)BSZ * NHEAD * 1024 * MCOND;
  float* f   = att;                                        // alias: free after attention
  float* u   = W + off; off += (size_t)N_NODES * 4096;

  hipMemsetAsync(colsum, 0, 1024 * sizeof(float), stream);
  hipMemsetAsync(deg, 0, N_NODES * sizeof(float), stream);
  hipMemsetAsync(h, 0, ND * sizeof(float), stream);

  // BatchNorm (training-mode batch stats, biased var)
  bn_stats_kernel<<<256, 256, 0, stream>>>(x, colsum, colsq);
  bn_apply_kernel<<<ND / 256, 256, 0, stream>>>(x, colsum, colsq, bn_w, bn_b, t0);

  // degrees + D^-1/2 (with self-loops)
  degree_kernel<<<(NEDGE + 255) / 256, 256, 0, stream>>>(ei, deg);
  dinv_kernel<<<(N_NODES + 255) / 256, 256, 0, stream>>>(deg, dinv);

  // GCN-in: h = scatter(bn(x) @ W) + b
  gemm(stream, false, t0, gcn_in_w, t1, nullptr, nullptr,
       N_NODES, DMODEL, DMODEL, DMODEL, DMODEL, DMODEL);
  gcn_scatter_kernel<<<NEDGE + N_NODES, 256, 0, stream>>>(t1, ei, dinv, h);
  add_bias_kernel<<<ND / 256, 256, 0, stream>>>(h, gcn_in_b);

  const long long DD = (long long)DMODEL * DMODEL;
  const int NTOK = N_NODES / BSZ;  // 1024 tokens per batch
  for (int l = 0; l < NLAYER; ++l) {
    // ---- cross-attention ----
    ln_kernel<<<N_NODES / 8, 256, 0, stream>>>(h, ln2_w + l * DMODEL, ln2_b + l * DMODEL, t0);
    gemm(stream, false, t0, Wq + l * DD, q, nullptr, nullptr,
         N_NODES, DMODEL, DMODEL, DMODEL, DMODEL, DMODEL);
    gemm(stream, false, cond, Wk + l * DD, kb, nullptr, nullptr,
         BSZ * MCOND, DMODEL, DMODEL, DMODEL, DMODEL, DMODEL);
    gemm(stream, false, cond, Wv + l * DD, vb, nullptr, nullptr,
         BSZ * MCOND, DMODEL, DMODEL, DMODEL, DMODEL, DMODEL);
    // att[b,h] = q[b,:,h,:] @ k[b,:,h,:]^T   (batched over 16x8)
    gemm(stream, true, q, kb, att, nullptr, nullptr,
         NTOK, MCOND, DHEAD, DMODEL, DMODEL, MCOND,
         BSZ, NHEAD,
         (long long)NTOK * DMODEL, DHEAD,
         (long long)MCOND * DMODEL, DHEAD,
         (long long)NHEAD * NTOK * MCOND, (long long)NTOK * MCOND);
    softmax_kernel<<<(BSZ * NHEAD * NTOK) / 8, 256, 0, stream>>>(att);
    // o[b,h] = att[b,h] @ v[b,:,h,:] -> t1 in [b*1024+m][h*64+d] layout
    gemm(stream, false, att, vb, t1, nullptr, nullptr,
         NTOK, DHEAD, MCOND, MCOND, DMODEL, DMODEL,
         BSZ, NHEAD,
         (long long)NHEAD * NTOK * MCOND, (long long)NTOK * MCOND,
         (long long)MCOND * DMODEL, DHEAD,
         (long long)NTOK * DMODEL, DHEAD);
    // h = h + o @ Wo
    gemm(stream, false, t1, Wo + l * DD, h, nullptr, h,
         N_NODES, DMODEL, DMODEL, DMODEL, DMODEL, DMODEL);

    // ---- GEGLU FFN ----
    ln_kernel<<<N_NODES / 8, 256, 0, stream>>>(h, ln3_w + l * DMODEL, ln3_b + l * DMODEL, t0);
    gemm(stream, false, t0, geglu_w + (long long)l * DMODEL * 4096, u,
         geglu_b + (long long)l * 4096, nullptr,
         N_NODES, 4096, DMODEL, DMODEL, 4096, 4096);
    geglu_kernel<<<((size_t)N_NODES * 2048) / 256, 256, 0, stream>>>(u, f);
    gemm(stream, false, f, ffo_w + (long long)l * 2048 * DMODEL, h,
         ffo_b + l * DMODEL, h,
         N_NODES, DMODEL, 2048, 2048, DMODEL, DMODEL);
  }

  // GCN-out + global residual: out = x + scatter(h @ W) + b
  gemm(stream, false, h, gcn_out_w, t0, nullptr, nullptr,
       N_NODES, DMODEL, DMODEL, DMODEL, DMODEL, DMODEL);
  hipMemcpyAsync(out, x, ND * sizeof(float), hipMemcpyDeviceToDevice, stream);
  gcn_scatter_kernel<<<NEDGE + N_NODES, 256, 0, stream>>>(t0, ei, dinv, out);
  add_bias_kernel<<<ND / 256, 256, 0, stream>>>(out, gcn_out_b);
}
